// genericPINN_53695681134759
// MI455X (gfx1250) — matrix operand order repaired and checked by reading, not compiled, not run on hardware
//
#include <hip/hip_runtime.h>
#include <hip/hip_bf16.h>

// MI455X / gfx1250, wave32, WMMA bf16 16x16x32 with f32 accumulate.

typedef __attribute__((ext_vector_type(16))) __bf16 v16bf;
typedef __attribute__((ext_vector_type(8)))  float  v8f;

#define N_LAYERS 8

__device__ __forceinline__ unsigned short f2bf(float f) {
  union { float f; unsigned u; } c; c.f = f;
  unsigned u = c.u;
  u += 0x7FFFu + ((u >> 16) & 1u);   // round-to-nearest-even
  return (unsigned short)(u >> 16);
}

// A-operand (16-bit, 16x32) element placement per ISA table:
// r = K%32:  [0,8): lane=m, e=r   [8,16): lane=m+16, e=r-8
//            [16,24): lane=m, e=8+(r-16)   [24,32): lane=m+16, e=8+(r-24)
__device__ __forceinline__ int a_unit(int buf, int smt, int kt, int mlocal, int r) {
  int lane = mlocal + (((r >> 3) & 1) << 4);
  int e    = (r & 7) + (((r >> 4) & 1) << 3);
  return (((buf * 8 + smt) * 16 + kt) << 9) + (lane << 4) + e;  // ushort units
}

// ---- Pre-pass: pack Ws (f32 row-major 512x512 per layer) into WMMA B-operand
// layout, bf16.  Layout: [layer][nTile 32][kTile 16][lane 32][e 16].
// B-operand (16-bit, 32x16): lane%16 = N, e<8: K=8*(lane/16)+e, e>=8: K=16+8*(lane/16)+(e-8).
__global__ void pack_w_kernel(const float* __restrict__ Ws,
                              unsigned short* __restrict__ out) {
  int o = blockIdx.x * 256 + threadIdx.x;        // 0 .. 2^21-1
  int e    =  o        & 15;
  int lane = (o >> 4)  & 31;
  int kt   = (o >> 9)  & 15;
  int n    = (o >> 13) & 31;
  int l    =  o >> 18;
  int K = kt * 32 + (e & 7) + ((lane >> 4) << 3) + ((e >> 3) << 4);
  int N = n * 16 + (lane & 15);
  out[o] = f2bf(Ws[(size_t)l * 262144 + K * 512 + N]);
}

// Pack Wf (512x4, f32) * scale, zero-padded to 16 columns, B-operand layout.
__global__ void pack_wf_kernel(const float* __restrict__ Wf,
                               const float* __restrict__ scale,
                               unsigned short* __restrict__ out) {
  int o = blockIdx.x * 256 + threadIdx.x;        // 0 .. 8191
  int e    =  o       & 15;
  int lane = (o >> 4) & 31;
  int kt   =  o >> 9;
  int K = kt * 32 + (e & 7) + ((lane >> 4) << 3) + ((e >> 3) << 4);
  int N = lane & 15;
  float v = (N < 4) ? Wf[K * 4 + N] * scale[N] : 0.0f;
  out[o] = f2bf(v);
}

// ---- Fused kernel: RFF front, 8 layers (primal + 3 tangents), final layer + curl.
// One workgroup = 256 threads (8 waves) handles 32 batch rows.
// LDS: two ping-pong activation buffers, each 4 states x 2 mtiles x 16 ktiles x 1KB.
// Each wave processes TWO n-columns at once so each A tile feeds 2 WMMAs.
__global__ __launch_bounds__(256, 1) void pinn_kernel(
    const float* __restrict__ inputs, const float* __restrict__ Brff,
    const float* __restrict__ bs,     const float* __restrict__ alphas,
    const unsigned short* __restrict__ ws, float* __restrict__ out) {
  __shared__ unsigned short lds[131072];   // 256 KB
  __shared__ float xs[128];
  const int tid  = threadIdx.x;
  const int lane = tid & 31;
  const int wid  = tid >> 5;
  const unsigned short* wW = ws;            // packed weights
  const unsigned short* wF = ws + 2097152;  // packed final layer

  if (tid < 128) xs[tid] = inputs[blockIdx.x * 128 + tid];
  __syncthreads();

  // ---- Front: u = [sin(x@B), cos(x@B)];  du_c = [cos(z)*B[c], -sin(z)*B[c]]
  for (int idx = tid; idx < 32 * 256; idx += 256) {
    int m = idx >> 8, j = idx & 255;
    float z = xs[m*4+0]*Brff[j]       + xs[m*4+1]*Brff[256+j]
            + xs[m*4+2]*Brff[512+j]   + xs[m*4+3]*Brff[768+j];
    float sz = __sinf(z), cz = __cosf(z);
    int mt = m >> 4, ml = m & 15;
    int K1 = j, K2 = j + 256;
    lds[a_unit(0, 0*2+mt, K1 >> 5, ml, K1 & 31)] = f2bf(sz);
    lds[a_unit(0, 0*2+mt, K2 >> 5, ml, K2 & 31)] = f2bf(cz);
#pragma unroll
    for (int c = 1; c < 4; ++c) {
      float bc = Brff[c*256 + j];
      lds[a_unit(0, c*2+mt, K1 >> 5, ml, K1 & 31)] = f2bf(cz * bc);
      lds[a_unit(0, c*2+mt, K2 >> 5, ml, K2 & 31)] = f2bf(-sz * bc);
    }
  }
  __syncthreads();

  // ---- Hidden layers
  int cur = 0, nxt = 1;
  for (int l = 0; l < N_LAYERS; ++l) {
    const float cc = 1.0f + alphas[l];
    const unsigned short* wL = wW + (size_t)l * 262144;
    for (int ni = 0; ni < 2; ++ni) {
      // two n-tiles per pass: n0 = wid + 16*ni, n1 = n0 + 8
      const int n0 = wid + ni * 16;
      float bval[2];
      bval[0] = bs[l * 512 + n0 * 16 + (lane & 15)];
      bval[1] = bs[l * 512 + (n0 + 8) * 16 + (lane & 15)];
      v8f acc[2][2][4];    // [col][mt][state]
      const v8f vzero = {0.f,0.f,0.f,0.f,0.f,0.f,0.f,0.f};
#pragma unroll
      for (int c = 0; c < 2; ++c)
#pragma unroll
        for (int mt = 0; mt < 2; ++mt)
#pragma unroll
          for (int s = 0; s < 4; ++s) acc[c][mt][s] = vzero;

      for (int kt = 0; kt < 16; ++kt) {
        v16bf b0 = *(const v16bf*)(wL + (((n0 * 16 + kt) << 9) + (lane << 4)));
        v16bf b1 = *(const v16bf*)(wL + ((((n0 + 8) * 16 + kt) << 9) + (lane << 4)));
#pragma unroll
        for (int mt = 0; mt < 2; ++mt)
#pragma unroll
          for (int s = 0; s < 4; ++s) {
            v16bf amat = *(const v16bf*)(
                &lds[(((cur * 8 + (s * 2 + mt)) * 16 + kt) << 9) + (lane << 4)]);
            acc[0][mt][s] = __builtin_amdgcn_wmma_f32_16x16x32_bf16(
                false, amat, false, b0, (short)0, acc[0][mt][s], false, false);
            acc[1][mt][s] = __builtin_amdgcn_wmma_f32_16x16x32_bf16(
                false, amat, false, b1, (short)0, acc[1][mt][s], false, false);
          }
      }

      // Epilogue: pre = acc0 + b; primal = sin(c*pre); tangent *= c*cos(c*pre)
      const int hi = lane >> 4;
#pragma unroll
      for (int c = 0; c < 2; ++c) {
        const int n = n0 + c * 8;
        const int K  = n * 16 + (lane & 15);
        const int kt2 = K >> 5, r = K & 31;
        const int lanebit = (r >> 3) & 1;
        const int e2 = (r & 7) + (((r >> 4) & 1) << 3);
#pragma unroll
        for (int mt = 0; mt < 2; ++mt) {
#pragma unroll
          for (int v = 0; v < 8; ++v) {
            float pre = acc[c][mt][0][v] + bval[c];
            float t  = cc * pre;
            float sv = __sinf(t);
            float gv = cc * __cosf(t);
            int ml2   = v + (hi << 3);
            int base  = ((ml2 + (lanebit << 4)) << 4) + e2;
            lds[(((nxt * 8 + (0 * 2 + mt)) * 16 + kt2) << 9) + base] = f2bf(sv);
#pragma unroll
            for (int s = 1; s < 4; ++s)
              lds[(((nxt * 8 + (s * 2 + mt)) * 16 + kt2) << 9) + base] =
                  f2bf(gv * acc[c][mt][s][v]);
          }
        }
      }
    }
    __syncthreads();
    cur ^= 1; nxt ^= 1;
  }

  // ---- Final layer: dP_c = du_c @ (Wf*scale), staged as f32 in the spare buffer
  float* stage = (float*)&lds[nxt * 65536];
  if (wid < 6) {
    const int s  = 1 + (wid % 3);   // tangent state
    const int mt = wid / 3;
    const v8f vzero = {0.f,0.f,0.f,0.f,0.f,0.f,0.f,0.f};
    v8f acc = vzero;
    for (int kt = 0; kt < 16; ++kt) {
      v16bf bmat = *(const v16bf*)(wF + ((kt * 32 + lane) << 4));
      v16bf amat = *(const v16bf*)(
          &lds[(((cur * 8 + (s * 2 + mt)) * 16 + kt) << 9) + (lane << 4)]);
      acc = __builtin_amdgcn_wmma_f32_16x16x32_bf16(
          false, amat, false, bmat, (short)0, acc, false, false);
    }
    const int Nl = lane & 15, hi = lane >> 4;
#pragma unroll
    for (int v = 0; v < 8; ++v)
      stage[((s - 1) * 2 + mt) * 256 + (v + (hi << 3)) * 16 + Nl] = acc[v];
  }
  __syncthreads();

  // ---- Curl combination: state1=d/dz, state2=d/dx, state3=d/dy
  if (tid < 96) {
    int m = tid / 3, comp = tid % 3;
    int mt = m >> 4, mm = m & 15;
#define DP(c, nn) stage[((c - 1) * 2 + mt) * 256 + mm * 16 + (nn)]
    float val;
    if (comp == 0)      val = DP(3, 2) - DP(1, 1) + DP(2, 3);  // u
    else if (comp == 1) val = DP(1, 0) - DP(2, 2) + DP(3, 3);  // v
    else                val = DP(2, 1) - DP(3, 0) + DP(1, 3);  // w
#undef DP
    out[(blockIdx.x * 32 + m) * 3 + comp] = val;
  }
}

extern "C" void kernel_launch(void* const* d_in, const int* in_sizes, int n_in,
                              void* d_out, int out_size, void* d_ws, size_t ws_size,
                              hipStream_t stream) {
  const float* inputs = (const float*)d_in[0];  // (65536,4)
  const float* Brff   = (const float*)d_in[1];  // (4,256)
  const float* Ws     = (const float*)d_in[2];  // (8,512,512)
  const float* bs     = (const float*)d_in[3];  // (8,512)
  const float* alphas = (const float*)d_in[4];  // (9,)
  const float* Wf     = (const float*)d_in[5];  // (512,4)
  const float* scale  = (const float*)d_in[6];  // (4,)
  (void)in_sizes; (void)n_in; (void)out_size; (void)ws_size;

  unsigned short* wpack = (unsigned short*)d_ws;  // 4 MB + 16 KB used
  pack_w_kernel <<<8192, 256, 0, stream>>>(Ws, wpack);
  pack_wf_kernel<<<32,   256, 0, stream>>>(Wf, scale, wpack + 2097152);
  pinn_kernel   <<<2048, 256, 0, stream>>>(inputs, Brff, bs, alphas, wpack,
                                           (float*)d_out);
}
